// Head_21053929685167
// MI455X (gfx1250) — compile-verified, matrix operand order checked by reading, not run on hardware
//
#include <hip/hip_runtime.h>
#include <hip/hip_bf16.h>
#include <stdint.h>

#define B_  16
#define T_  2048
#define E_  512
#define D_  64

typedef __attribute__((ext_vector_type(16))) __bf16 v16bf;
typedef __attribute__((ext_vector_type(8)))  float  v8f;
typedef __attribute__((ext_vector_type(4)))  int    v4i;

#if defined(__gfx1250__) && __has_builtin(__builtin_amdgcn_global_load_async_to_lds_b128)
#define HAVE_ASYNC_LDS 1
#endif

// Round two f32 to bf16 (RNE) and pack into one u32 via v_perm_b32.
static __device__ __forceinline__ uint32_t pkbf(float a, float b) {
  uint32_t ua = __builtin_bit_cast(uint32_t, a);
  uint32_t ub = __builtin_bit_cast(uint32_t, b);
  ua += 0x7FFFu + ((ua >> 16) & 1u);
  ub += 0x7FFFu + ((ub >> 16) & 1u);
  // out = {ub[31:24], ub[23:16], ua[31:24], ua[23:16]}
  return __builtin_amdgcn_perm(ub, ua, 0x07060302u);
}
static __device__ __forceinline__ uint16_t f2bfu(float f) {
  uint32_t u = __builtin_bit_cast(uint32_t, f);
  u += 0x7FFFu + ((u >> 16) & 1u);
  return (uint16_t)(u >> 16);
}
static __device__ __forceinline__ float bf2f(uint16_t h) {
  return __builtin_bit_cast(float, (uint32_t)h << 16);   // exact
}

static __device__ __forceinline__ v8f wmma_bf16(v16bf a, v16bf b, v8f c) {
  // D = A(16x32 bf16) * B(32x16 bf16) + C(16x16 f32)
  return __builtin_amdgcn_wmma_f32_16x16x32_bf16(false, a, false, b, (short)0, c,
                                                 false, false);
}

// Copy 16 bytes global -> LDS (async path if available)
static __device__ __forceinline__ void stage16(const uint16_t* g, uint16_t* l) {
#if HAVE_ASYNC_LDS
  typedef __attribute__((address_space(1))) v4i* gv4i_p;
  typedef __attribute__((address_space(3))) v4i* lv4i_p;
  __builtin_amdgcn_global_load_async_to_lds_b128((gv4i_p)g, (lv4i_p)l, 0, 0);
#else
  *reinterpret_cast<uint4*>(l) = *reinterpret_cast<const uint4*>(g);
#endif
}
static __device__ __forceinline__ void stage_wait() {
#if HAVE_ASYNC_LDS
#if __has_builtin(__builtin_amdgcn_s_wait_asynccnt)
  __builtin_amdgcn_s_wait_asynccnt(0);
#else
  asm volatile("s_wait_asynccnt 0" ::: "memory");
#endif
#endif
}

// ---------------------------------------------------------------------------
// Kernel 0: WqT[n][k] = bf16(Wq[k][n])  (64x512 bf16, 64KB -> stays L2 hot)
// ---------------------------------------------------------------------------
__global__ __launch_bounds__(256)
void k_wt(const float* __restrict__ Wq, uint16_t* __restrict__ WqT)
{
  const int idx = blockIdx.x * 256 + threadIdx.x;      // < 32768
  const int k = idx & (E_ - 1);
  const int n = idx >> 9;
  WqT[(size_t)n * E_ + k] = f2bfu(Wq[(size_t)k * D_ + n]);
}

// ---------------------------------------------------------------------------
// Kernel 1: q = x @ Wq + bq (bf16 out). One wave per 16x64 q tile; B from WqT
// (global, contiguous v16bf per lane, L2 resident). No LDS needed.
// ---------------------------------------------------------------------------
__global__ __launch_bounds__(256)
void k_proj(const float* __restrict__ x, const uint16_t* __restrict__ WqT,
            const float* __restrict__ bq, uint16_t* __restrict__ qb)
{
  const int tid  = threadIdx.x;
  const int wave = tid >> 5;
  const int lane = tid & 31;
  const int n    = lane & 15;      // A-layout M / C-layout N
  const int hi   = lane >> 4;      // K-half selector
  const int tok0 = blockIdx.x * 128 + wave * 16;

  v8f acc[4] = {v8f{}, v8f{}, v8f{}, v8f{}};
  const float* xrow = x + (size_t)(tok0 + n) * E_;

  for (int kb = 0; kb < E_; kb += 32) {
    if (kb + 32 < E_)
      __builtin_prefetch(xrow + kb + 32 + hi * 16, 0, 0);

    // A operand: x[tok0+n][kb + hi*16 .. +15] -> bf16 (perm-packed converts)
    const float4* xr = reinterpret_cast<const float4*>(xrow + kb + hi * 16);
    float4 p0 = xr[0], p1 = xr[1], p2 = xr[2], p3 = xr[3];
    float fv[16] = {p0.x, p0.y, p0.z, p0.w, p1.x, p1.y, p1.z, p1.w,
                    p2.x, p2.y, p2.z, p2.w, p3.x, p3.y, p3.z, p3.w};
    union { v16bf v; uint32_t w[8]; } ua;
#pragma unroll
    for (int q = 0; q < 8; ++q) ua.w[q] = pkbf(fv[2 * q], fv[2 * q + 1]);

#pragma unroll
    for (int nt = 0; nt < 4; ++nt) {
      v16bf bv = *reinterpret_cast<const v16bf*>(
          WqT + (size_t)(nt * 16 + n) * E_ + kb + hi * 16);
      acc[nt] = wmma_bf16(ua.v, bv, acc[nt]);
    }
  }

  // Epilogue: C layout -> lane holds column n, rows r + hi*8
#pragma unroll
  for (int nt = 0; nt < 4; ++nt) {
    const int col  = nt * 16 + n;
    const float bb = bq[col];
#pragma unroll
    for (int r = 0; r < 8; ++r) {
      const size_t row = (size_t)tok0 + r + hi * 8;
      qb[row * D_ + col] = f2bfu(acc[nt][r] + bb);
    }
  }
}

// ---------------------------------------------------------------------------
// Kernel 2: column softmax stats: m_j = max_i S_ij, Z_j = sum_i e^{S_ij-m_j}.
// 8 waves/block share batch b; a 32-row i-chunk of q (4KB) is staged per block
// via async global->LDS, then all waves run 2 sub-tiles (4 WMMAs) per stage.
// ---------------------------------------------------------------------------
__global__ __launch_bounds__(256)
void k_stats(const uint16_t* __restrict__ qb, float* __restrict__ mj,
             float* __restrict__ zj)
{
  __shared__ __align__(32) uint16_t sA[32 * D_];       // 4KB: current i-chunk
  const int tid = threadIdx.x, wave = tid >> 5, lane = tid & 31;
  const int gw = blockIdx.x * 8 + wave;                // 0..2047
  const int b  = gw >> 7;
  const int j0 = (gw & 127) * 16;
  const int n  = lane & 15, hi = lane >> 4;
  const size_t bT = (size_t)b * T_;
  const uint16_t* qbase = qb + bT * D_;

  // B operand fixed over i loop: B[k][n] = q[j0+n][k]
  const uint16_t* rowB = qbase + (size_t)(j0 + n) * D_;
  v16bf b0 = *reinterpret_cast<const v16bf*>(rowB + hi * 16);
  v16bf b1 = *reinterpret_cast<const v16bf*>(rowB + 32 + hi * 16);

  float mloc = -1.0e30f, zloc = 0.0f;
  for (int i0 = 0; i0 < T_; i0 += 32) {
    // stage q rows i0..i0+31 (4KB = 256 x 16B, one async chunk per thread)
    stage16(qbase + (size_t)i0 * D_ + tid * 8, sA + tid * 8);
    stage_wait();
    __syncthreads();

    v8f c0 = {}, c1 = {};
    {
      const uint16_t* ar = sA + (size_t)n * D_;
      v16bf a0 = *reinterpret_cast<const v16bf*>(ar + hi * 16);
      v16bf a1 = *reinterpret_cast<const v16bf*>(ar + 32 + hi * 16);
      c0 = wmma_bf16(a0, b0, c0);
      c0 = wmma_bf16(a1, b1, c0);
    }
    {
      const uint16_t* ar = sA + (size_t)(16 + n) * D_;
      v16bf a0 = *reinterpret_cast<const v16bf*>(ar + hi * 16);
      v16bf a1 = *reinterpret_cast<const v16bf*>(ar + 32 + hi * 16);
      c1 = wmma_bf16(a0, b0, c1);
      c1 = wmma_bf16(a1, b1, c1);
    }

    float tmax = -1.0e30f;
#pragma unroll
    for (int r = 0; r < 8; ++r) {
      tmax = fmaxf(tmax, c0[r] * 0.125f);
      tmax = fmaxf(tmax, c1[r] * 0.125f);
    }
    const float nm = fmaxf(mloc, tmax);
    float zs = zloc * __expf(mloc - nm);
#pragma unroll
    for (int r = 0; r < 8; ++r) {
      zs += __expf(c0[r] * 0.125f - nm);
      zs += __expf(c1[r] * 0.125f - nm);
    }
    mloc = nm; zloc = zs;
    __syncthreads();                                   // before next stage
  }

  // combine lane pair (l, l^16): each holds 8 of the 16 rows of column n
  const float mo = __shfl_xor(mloc, 16, 32);
  const float zo = __shfl_xor(zloc, 16, 32);
  const float M  = fmaxf(mloc, mo);
  const float Z  = zloc * __expf(mloc - M) + zo * __expf(mo - M);
  if (lane < 16) {
    mj[bT + j0 + lane] = M;
    zj[bT + j0 + lane] = Z;
  }
}

// ---------------------------------------------------------------------------
// Kernel 3: q~^T[b][d][t] = bf16(q[b,t,d] / Z_{b,t})   (transposed layout so
// the P x q~ B operand is a contiguous global v16bf load)
// ---------------------------------------------------------------------------
__global__ __launch_bounds__(256)
void k_qzt(const uint16_t* __restrict__ qb, const float* __restrict__ zj,
           uint16_t* __restrict__ qzT)
{
  const size_t p = (size_t)blockIdx.x * 256 + threadIdx.x;  // < B*D*T/2
  const int t = (int)(p & 1023) * 2;
  const int d = (int)(p >> 10) & 63;
  const int b = (int)(p >> 16);
  const size_t r0 = (size_t)b * T_ + t;
  const float z0 = zj[r0], z1 = zj[r0 + 1];
  const float f0 = bf2f(qb[r0 * D_ + d]);
  const float f1 = bf2f(qb[(r0 + 1) * D_ + d]);
  *reinterpret_cast<uint32_t*>(qzT + ((size_t)b * D_ + d) * T_ + t) =
      pkbf(f0 / z0, f1 / z1);
}

// ---------------------------------------------------------------------------
// Kernel 4: out[i,:] = sum_j exp(S_ij/8 - m_j) * q~[j,:]
// 8 waves/block share batch b; the j-chunk of q (4KB) is staged per block via
// async global->LDS. P tile transposed (C->A layout) through per-wave LDS.
// B operand of P x q~ read directly from qzT (global, contiguous).
// ---------------------------------------------------------------------------
#define PSTR 48
__global__ __launch_bounds__(256)
void k_attn(const uint16_t* __restrict__ qb, const uint16_t* __restrict__ qzT,
            const float* __restrict__ mj, float* __restrict__ out)
{
  __shared__ __align__(32) uint16_t sQ[32 * D_];        // 4KB shared j-chunk
  __shared__ __align__(32) uint16_t sP[8][16 * PSTR];   // per-wave P tiles
  const int tid = threadIdx.x, wave = tid >> 5, lane = tid & 31;
  const int gw = blockIdx.x * 8 + wave;                 // 0..2047
  const int b  = gw >> 7;
  const int i0 = (gw & 127) * 16;
  const int n  = lane & 15, hi = lane >> 4;
  const size_t bT = (size_t)b * T_;
  uint16_t* P = sP[wave];
  const uint16_t* qbase = qb + bT * D_;
  const uint16_t* zbase = qzT + (size_t)b * D_ * T_;

  // A operand for S: rows i0..i0+15 of q (fixed over j loop)
  const uint16_t* rowA = qbase + (size_t)(i0 + n) * D_;
  v16bf a0 = *reinterpret_cast<const v16bf*>(rowA + hi * 16);
  v16bf a1 = *reinterpret_cast<const v16bf*>(rowA + 32 + hi * 16);

  v8f acc[4] = {v8f{}, v8f{}, v8f{}, v8f{}};

  for (int j0 = 0; j0 < T_; j0 += 32) {
    // stage q rows j0..j0+31 (4KB = 256 x 16B, one chunk per thread)
    stage16(qbase + (size_t)j0 * D_ + tid * 8, sQ + tid * 8);
    stage_wait();
    __syncthreads();

    // two 16x16 S sub-tiles -> P = exp(S/8 - m_j), bf16 row-major in LDS
#pragma unroll
    for (int h = 0; h < 2; ++h) {
      const uint16_t* rb = sQ + (size_t)(h * 16 + n) * D_;
      v16bf bb0 = *reinterpret_cast<const v16bf*>(rb + hi * 16);
      v16bf bb1 = *reinterpret_cast<const v16bf*>(rb + 32 + hi * 16);
      v8f s = {};
      s = wmma_bf16(a0, bb0, s);
      s = wmma_bf16(a1, bb1, s);
      const float mv = mj[bT + j0 + h * 16 + n];
#pragma unroll
      for (int r = 0; r < 8; ++r)
        P[(r + hi * 8) * PSTR + h * 16 + n] = f2bfu(__expf(s[r] * 0.125f - mv));
    }
    __syncthreads();   // P writes (cross-lane transpose) -> reads

    // A = P (16x32), B = q~^T rows (global, contiguous per lane)
    v16bf pa = *reinterpret_cast<const v16bf*>(&P[n * PSTR + hi * 16]);
#pragma unroll
    for (int nt = 0; nt < 4; ++nt) {
      v16bf bv = *reinterpret_cast<const v16bf*>(
          zbase + (size_t)(nt * 16 + n) * T_ + j0 + hi * 16);
      acc[nt] = wmma_bf16(pa, bv, acc[nt]);
    }
    __syncthreads();   // reads done before next iteration overwrites sQ/sP
  }

#pragma unroll
  for (int nt = 0; nt < 4; ++nt)
#pragma unroll
    for (int r = 0; r < 8; ++r)
      out[(bT + i0 + r + hi * 8) * D_ + nt * 16 + n] = acc[nt][r];
}

// ---------------------------------------------------------------------------
extern "C" void kernel_launch(void* const* d_in, const int* in_sizes, int n_in,
                              void* d_out, int out_size, void* d_ws, size_t ws_size,
                              hipStream_t stream)
{
  (void)in_sizes; (void)n_in; (void)out_size; (void)ws_size;
  const float* x  = (const float*)d_in[0];
  const float* Wq = (const float*)d_in[1];
  const float* bq = (const float*)d_in[2];
  float* out = (float*)d_out;

  char* ws = (char*)d_ws;
  uint16_t* qb  = (uint16_t*)(ws);                                 // 4 MB  q bf16
  uint16_t* qzT = (uint16_t*)(ws + ((size_t)4 << 20));             // 4 MB  (q/Z)^T bf16
  uint16_t* WqT = (uint16_t*)(ws + ((size_t)8 << 20));             // 64 KB Wq^T bf16
  float*    mj  = (float*)   (ws + ((size_t)9 << 20));             // 128 KB col max
  float*    zj  = (float*)   (ws + ((size_t)9 << 20) + (1 << 17)); // 128 KB col sum

  k_wt   <<<128,  256, 0, stream>>>(Wq, WqT);
  k_proj <<<256,  256, 0, stream>>>(x, WqT, bq, qb);
  k_stats<<<256,  256, 0, stream>>>(qb, mj, zj);
  k_qzt  <<<4096, 256, 0, stream>>>(qb, zj, qzT);
  k_attn <<<256,  256, 0, stream>>>(qb, qzT, mj, out);
}
